// FC_9466107920597
// MI455X (gfx1250) — compile-verified, matrix operand order checked by reading, not run on hardware
//
#include <hip/hip_runtime.h>
#include <hip/hip_bf16.h>
#include <stdint.h>

typedef __attribute__((ext_vector_type(16))) __bf16 v16bf;
typedef __attribute__((ext_vector_type(8)))  float  v8f;

#define EPS_BN 1e-5f
#define SLOPE  0.01f

__device__ __forceinline__ uint16_t f2bf_bits(float f) {
  uint32_t u = __builtin_bit_cast(uint32_t, f);
  uint32_t r = (u + 0x7FFFu + ((u >> 16) & 1u)) >> 16;   // round-to-nearest-even
  return (uint16_t)r;
}
__device__ __forceinline__ __bf16 f2bf(float f) {
  uint16_t h = f2bf_bits(f);
  return __builtin_bit_cast(__bf16, h);
}
__device__ __forceinline__ uint32_t pack2(float lo, float hi) {
  return (uint32_t)f2bf_bits(lo) | ((uint32_t)f2bf_bits(hi) << 16);
}
__device__ __forceinline__ float leaky(float x) { return x >= 0.f ? x : SLOPE * x; }

// CDNA5 async global->LDS copy (ASYNCcnt); lds_off = byte offset in LDS space
__device__ __forceinline__ void async_cp16(uint32_t lds_off, const __bf16* g) {
  uint64_t ga = (uint64_t)(uintptr_t)g;
  asm volatile("global_load_async_to_lds_b128 %0, %1, off"
               :: "v"(lds_off), "v"(ga) : "memory");
}
__device__ __forceinline__ void wait_async0() {
  asm volatile("s_wait_asynccnt 0x0" ::: "memory");
}

// ---------------------------------------------------------------------------
// Operand prep: zero-padded bf16 buffers (Kp mult 32, Mp mult 128, Np mult 128)
// ---------------------------------------------------------------------------
__global__ void convert_pad_a_kernel(const float* __restrict__ src, __bf16* __restrict__ dst,
                                     int M, int K, int Mp, int Kp) {
  const int ck = Kp >> 3;
  size_t idx = (size_t)blockIdx.x * blockDim.x + threadIdx.x;
  if (idx >= (size_t)Mp * ck) return;
  const int row = (int)(idx / ck);
  const int k   = (int)(idx % ck) * 8;
  float v[8];
  if (row < M && k + 8 <= K) {
    const float* s = src + (size_t)row * K + k;
    float4 a = ((const float4*)s)[0];
    float4 b = ((const float4*)s)[1];
    v[0]=a.x; v[1]=a.y; v[2]=a.z; v[3]=a.w; v[4]=b.x; v[5]=b.y; v[6]=b.z; v[7]=b.w;
  } else {
    #pragma unroll
    for (int c = 0; c < 8; ++c)
      v[c] = (row < M && k + c < K) ? src[(size_t)row * K + k + c] : 0.f;
  }
  uint4 o = { pack2(v[0],v[1]), pack2(v[2],v[3]), pack2(v[4],v[5]), pack2(v[6],v[7]) };
  *(uint4*)(dst + (size_t)row * Kp + k) = o;
}

__global__ void transpose_convert_w_kernel(const float* __restrict__ src, __bf16* __restrict__ dst,
                                           int K, int N, int Np, int Kp) {
  const int ck = Kp >> 3;
  size_t idx = (size_t)blockIdx.x * blockDim.x + threadIdx.x;
  if (idx >= (size_t)Np * ck) return;
  const int n = (int)(idx / ck);
  const int k = (int)(idx % ck) * 8;
  float v[8];
  #pragma unroll
  for (int c = 0; c < 8; ++c)
    v[c] = (n < N && k + c < K) ? src[(size_t)(k + c) * N + n] : 0.f;
  uint4 o = { pack2(v[0],v[1]), pack2(v[2],v[3]), pack2(v[4],v[5]), pack2(v[6],v[7]) };
  *(uint4*)(dst + (size_t)n * Kp + k) = o;
}

// ---------------------------------------------------------------------------
// GEMM: C = A(bf16 [Mp,lda]) x Bt(bf16 [Np,ldb], N-major), f32 accumulate.
// 128x128 block, 8 waves of 32x64 (2x4 WMMA tiles). Double-buffered LDS with
// async global->LDS staging overlapped with WMMA.
// epi: 0=none, 1=bias->BN->leaky (enc), 2=bias->leaky->BN (fc1)
// ---------------------------------------------------------------------------
#define BM 128
#define BN 128
#define BK 32
#define LDSS 40                 // bf16 per LDS row: 32 + 8 pad (80B rows)
#define TBUF (BM * LDSS * 2)    // bytes per A/B tile buffer (10240)

__global__ __launch_bounds__(256)
void gemm_bf16_wmma(const __bf16* __restrict__ A, int lda,
                    const __bf16* __restrict__ Bt, int ldb,
                    float* __restrict__ C32, int ldc32, int off32,
                    __bf16* __restrict__ C16, int ldc16, int off16,
                    int M, int N, int Kp, int epi,
                    const float* __restrict__ bias,
                    const float* __restrict__ gamma,
                    const float* __restrict__ beta,
                    const float* __restrict__ mean,
                    const float* __restrict__ var)
{
  __shared__ __bf16 sA[2 * BM * LDSS];
  __shared__ __bf16 sB[2 * BN * LDSS];

  const int tid  = threadIdx.x;
  const int lane = tid & 31;
  const int wave = tid >> 5;
  const int wm   = wave & 3;   // 4 waves over M (32 rows each)
  const int wn   = wave >> 2;  // 2 waves over N (64 cols each)
  const int m0   = blockIdx.y * BM;
  const int n0   = blockIdx.x * BN;

  const v8f vzero = {0.f,0.f,0.f,0.f,0.f,0.f,0.f,0.f};
  v8f acc[2][4];
  #pragma unroll
  for (int i = 0; i < 2; ++i)
    #pragma unroll
    for (int j = 0; j < 4; ++j) acc[i][j] = vzero;

  // async staging: each thread copies 32B of A and 32B of B per k-step
  const __bf16* gA = A  + (size_t)(m0 + (tid >> 1)) * lda + (tid & 1) * 16;
  const __bf16* gB = Bt + (size_t)(n0 + (tid >> 1)) * ldb + (tid & 1) * 16;
  const uint32_t ldsA = (uint32_t)(size_t)&sA[(tid >> 1) * LDSS + (tid & 1) * 16];
  const uint32_t ldsB = (uint32_t)(size_t)&sB[(tid >> 1) * LDSS + (tid & 1) * 16];

  const int mlane = lane & 15;
  const int glane = lane >> 4;
  const int fA0 = (wm * 32 +  0 + mlane) * LDSS + glane * 8;
  const int fA1 = (wm * 32 + 16 + mlane) * LDSS + glane * 8;
  const int fB0 = (wn * 64 +  0 + mlane) * LDSS + glane * 8;
  const int fB1 = (wn * 64 + 16 + mlane) * LDSS + glane * 8;
  const int fB2 = (wn * 64 + 32 + mlane) * LDSS + glane * 8;
  const int fB3 = (wn * 64 + 48 + mlane) * LDSS + glane * 8;

  // prologue: stage tile 0 into buffer 0
  async_cp16(ldsA,      gA);
  async_cp16(ldsA + 16, gA + 8);
  async_cp16(ldsB,      gB);
  async_cp16(ldsB + 16, gB + 8);
  wait_async0();
  __syncthreads();

  int buf = 0;
  for (int k0 = 0; k0 < Kp; k0 += BK) {
    if (k0 + BK < Kp) {                       // stage next tile into other buffer
      const uint32_t boA = ldsA + (buf ^ 1) * TBUF;
      const uint32_t boB = ldsB + (buf ^ 1) * TBUF;
      async_cp16(boA,      gA + k0 + BK);
      async_cp16(boA + 16, gA + k0 + BK + 8);
      async_cp16(boB,      gB + k0 + BK);
      async_cp16(boB + 16, gB + k0 + BK + 8);
    }

    const __bf16* bA = sA + buf * (BM * LDSS);
    const __bf16* bB = sB + buf * (BN * LDSS);
    v16bf a0, a1, b0, b1, b2, b3;
    *((float4*)&a0)     = *(const float4*)(bA + fA0);
    *((float4*)&a0 + 1) = *(const float4*)(bA + fA0 + 16);
    *((float4*)&a1)     = *(const float4*)(bA + fA1);
    *((float4*)&a1 + 1) = *(const float4*)(bA + fA1 + 16);
    *((float4*)&b0)     = *(const float4*)(bB + fB0);
    *((float4*)&b0 + 1) = *(const float4*)(bB + fB0 + 16);
    *((float4*)&b1)     = *(const float4*)(bB + fB1);
    *((float4*)&b1 + 1) = *(const float4*)(bB + fB1 + 16);
    *((float4*)&b2)     = *(const float4*)(bB + fB2);
    *((float4*)&b2 + 1) = *(const float4*)(bB + fB2 + 16);
    *((float4*)&b3)     = *(const float4*)(bB + fB3);
    *((float4*)&b3 + 1) = *(const float4*)(bB + fB3 + 16);

    acc[0][0] = __builtin_amdgcn_wmma_f32_16x16x32_bf16(false, a0, false, b0, (short)0, acc[0][0], false, false);
    acc[0][1] = __builtin_amdgcn_wmma_f32_16x16x32_bf16(false, a0, false, b1, (short)0, acc[0][1], false, false);
    acc[0][2] = __builtin_amdgcn_wmma_f32_16x16x32_bf16(false, a0, false, b2, (short)0, acc[0][2], false, false);
    acc[0][3] = __builtin_amdgcn_wmma_f32_16x16x32_bf16(false, a0, false, b3, (short)0, acc[0][3], false, false);
    acc[1][0] = __builtin_amdgcn_wmma_f32_16x16x32_bf16(false, a1, false, b0, (short)0, acc[1][0], false, false);
    acc[1][1] = __builtin_amdgcn_wmma_f32_16x16x32_bf16(false, a1, false, b1, (short)0, acc[1][1], false, false);
    acc[1][2] = __builtin_amdgcn_wmma_f32_16x16x32_bf16(false, a1, false, b2, (short)0, acc[1][2], false, false);
    acc[1][3] = __builtin_amdgcn_wmma_f32_16x16x32_bf16(false, a1, false, b3, (short)0, acc[1][3], false, false);

    wait_async0();          // our async copies into buf^1 have landed
    __syncthreads();        // all waves done reading buf / writing buf^1
    buf ^= 1;
  }

  // Epilogue. C layout: VGPR j -> m_local = (lane>>4)*8 + j, n_local = lane&15.
  #pragma unroll
  for (int mt = 0; mt < 2; ++mt) {
    #pragma unroll
    for (int nt = 0; nt < 4; ++nt) {
      const int ncol = n0 + wn * 64 + nt * 16 + mlane;
      if (ncol >= N) continue;
      float bia = 0.f, ga = 0.f, be = 0.f, me = 0.f, iv = 0.f;
      if (epi) {
        bia = bias[ncol];
        ga = gamma[ncol]; be = beta[ncol]; me = mean[ncol];
        iv = rsqrtf(var[ncol] + EPS_BN);
      }
      #pragma unroll
      for (int j = 0; j < 8; ++j) {
        const int mrow = m0 + wm * 32 + mt * 16 + glane * 8 + j;
        if (mrow >= M) continue;
        float v = acc[mt][nt][j];
        if (epi == 1) { v = leaky(ga * (v + bia - me) * iv + be); }       // enc
        else if (epi == 2) { v = ga * (leaky(v + bia) - me) * iv + be; }  // fc1
        if (C32) C32[(size_t)mrow * ldc32 + off32 + ncol] = v;
        if (C16) C16[(size_t)mrow * ldc16 + off16 + ncol] = f2bf(v);
      }
    }
  }
}

// ---------------------------------------------------------------------------
// GCN normalization + aggregation
// ---------------------------------------------------------------------------
__global__ void set_val_kernel(float* __restrict__ p, float v, int n) {
  int i = blockIdx.x * blockDim.x + threadIdx.x;
  if (i < n) p[i] = v;
}

__global__ void edge_deg_kernel(const int* __restrict__ dst, const float* __restrict__ w,
                                float* __restrict__ deg, int E) {
  int e = blockIdx.x * blockDim.x + threadIdx.x;
  if (e < E) atomicAdd(&deg[dst[e]], w[e]);
}

__global__ void deg_to_dinv_kernel(float* __restrict__ deg, int n) {
  int i = blockIdx.x * blockDim.x + threadIdx.x;
  if (i < n) { float d = deg[i]; deg[i] = d > 0.f ? rsqrtf(d) : 0.f; }
}

__global__ void self_loop_init_kernel(const float* __restrict__ h, const float* __restrict__ dinv,
                                      float* __restrict__ out, int N, int F4) {
  size_t i = (size_t)blockIdx.x * blockDim.x + threadIdx.x;
  if (i >= (size_t)N * F4) return;
  int row = (int)(i / F4), f = (int)(i % F4);
  float s = dinv[row]; s = 2.f * s * s;
  float4 hv = ((const float4*)h)[(size_t)row * F4 + f];
  float4 o; o.x = s*hv.x; o.y = s*hv.y; o.z = s*hv.z; o.w = s*hv.w;
  ((float4*)out)[(size_t)row * F4 + f] = o;
}

__global__ void edge_aggregate_kernel(const int* __restrict__ src, const int* __restrict__ dst,
                                      const float* __restrict__ w, const float* __restrict__ dinv,
                                      const float* __restrict__ h, float* __restrict__ out, int F4) {
  const int e = blockIdx.x;
  const int s = src[e], d = dst[e];
  const float nrm = dinv[s] * w[e] * dinv[d];
  const float4* hv = (const float4*)(h + (size_t)s * (F4 * 4));
  float* op = out + (size_t)d * (F4 * 4);
  for (int f = threadIdx.x; f < F4; f += blockDim.x) {
    float4 v = hv[f];
    atomicAdd(&op[f*4+0], nrm * v.x);
    atomicAdd(&op[f*4+1], nrm * v.y);
    atomicAdd(&op[f*4+2], nrm * v.z);
    atomicAdd(&op[f*4+3], nrm * v.w);
  }
}

__global__ void gather_bias_leaky_bf16_kernel(const float* __restrict__ agg,
                                              const int* __restrict__ idx,
                                              const float* __restrict__ bias,
                                              __bf16* __restrict__ out,
                                              int B, int F, int ldo, int col_off) {
  size_t i = (size_t)blockIdx.x * blockDim.x + threadIdx.x;
  if (i >= (size_t)B * F) return;
  int b = (int)(i / F), f = (int)(i % F);
  float v = agg[(size_t)idx[b] * F + f] + bias[f];
  out[(size_t)b * ldo + col_off + f] = f2bf(leaky(v));
}

__global__ void copy_cols_bf16_kernel(const float* __restrict__ in, __bf16* __restrict__ out,
                                      int B, int F, int ldo, int col_off) {
  size_t i = (size_t)blockIdx.x * blockDim.x + threadIdx.x;
  if (i >= (size_t)B * F) return;
  int b = (int)(i / F), f = (int)(i % F);
  out[(size_t)b * ldo + col_off + f] = f2bf(in[(size_t)b * F + f]);
}

__global__ void zero_cols_bf16_kernel(__bf16* __restrict__ out, int B, int ldo,
                                      int col_off, int cnt) {
  size_t i = (size_t)blockIdx.x * blockDim.x + threadIdx.x;
  if (i >= (size_t)B * cnt) return;
  int b = (int)(i / cnt), f = (int)(i % cnt);
  out[(size_t)b * ldo + col_off + f] = f2bf(0.f);
}

__global__ void fc2_relu_kernel(const float* __restrict__ h, const float* __restrict__ W,
                                const float* __restrict__ b, float* __restrict__ y,
                                int B, int F) {
  int row  = blockIdx.x * (blockDim.x >> 5) + (threadIdx.x >> 5);
  int lane = threadIdx.x & 31;
  if (row >= B) return;
  const float* hp = h + (size_t)row * F;
  float s = 0.f;
  for (int f = lane; f < F; f += 32) s += hp[f] * W[f];
  #pragma unroll
  for (int off = 16; off; off >>= 1) s += __shfl_xor(s, off, 32);
  if (lane == 0) y[row] = fmaxf(s + b[0], 0.f);
}

// ---------------------------------------------------------------------------
static inline int ceilN(int x, int n) { return (x + n - 1) / n * n; }

extern "C" void kernel_launch(void* const* d_in, const int* in_sizes, int n_in,
                              void* d_out, int out_size, void* d_ws, size_t ws_size,
                              hipStream_t stream) {
  (void)in_sizes; (void)n_in; (void)out_size; (void)ws_size;
  const int* d_index = (const int*)d_in[0];
  const int* p_index = (const int*)d_in[1];
  const float* p_emb   = (const float*)d_in[2];
  const float* d_vecs  = (const float*)d_in[3];
  const float* d_ecfps = (const float*)d_in[4];
  const float* p_gos   = (const float*)d_in[5];
  const int*   d_ei    = (const int*)d_in[6];
  const float* d_ew    = (const float*)d_in[7];
  const int*   p_ei    = (const int*)d_in[8];
  const float* p_ew    = (const float*)d_in[9];
  const float* W_dvec  = (const float*)d_in[10]; const float* b_dvec  = (const float*)d_in[11];
  const float* W_decfp = (const float*)d_in[12]; const float* b_decfp = (const float*)d_in[13];
  const float* W_pgo   = (const float*)d_in[14]; const float* b_pgo   = (const float*)d_in[15];
  const float* enc_W   = (const float*)d_in[16]; const float* enc_b   = (const float*)d_in[17];
  const float* enc_g   = (const float*)d_in[18]; const float* enc_be  = (const float*)d_in[19];
  const float* enc_m   = (const float*)d_in[20]; const float* enc_v   = (const float*)d_in[21];
  const float* fc1_W   = (const float*)d_in[22]; const float* fc1_b   = (const float*)d_in[23];
  const float* fc_g    = (const float*)d_in[24]; const float* fc_be   = (const float*)d_in[25];
  const float* fc_m    = (const float*)d_in[26]; const float* fc_v    = (const float*)d_in[27];
  const float* fc2_W   = (const float*)d_in[28]; const float* fc2_b   = (const float*)d_in[29];

  const int B = 8192, ND = 10000, NP = 20000, ED = 320000, EP = 640000;

  char* ws = (char*)d_ws;
  size_t off = 0;
  auto alloc = [&](size_t bytes) { char* p = ws + off; off += (bytes + 255) & ~(size_t)255; return p; };
  float*  dinvd   = (float*)alloc(sizeof(float) * ND);
  float*  dinvp   = (float*)alloc(sizeof(float) * NP);
  float*  hbuf    = (float*)alloc(sizeof(float) * (size_t)NP * 1024);
  float*  aggbuf  = (float*)alloc(sizeof(float) * (size_t)NP * 1024);
  __bf16* abuf    = (__bf16*)alloc(2ull * 20096 * 2816);   // padded bf16 A (max)
  __bf16* wtbuf   = (__bf16*)alloc(2ull * 1024 * 2816);    // padded bf16 W^T (max)
  __bf16* enc_in  = (__bf16*)alloc(2ull * B * 1344);       // [dv | p_emb | pad]
  __bf16* fc1_in  = (__bf16*)alloc(2ull * B * 2560);       // [feature | ec | go]
  float*  fc1_out = (float*)alloc(sizeof(float) * (size_t)B * 512);
  float*  y_out   = (float*)d_out;                          // y[8192], then feature

  dim3 blk(256);
  auto g1 = [](size_t n) { return dim3((unsigned)((n + 255) / 256)); };

  // ---- degrees / symmetric normalization (improved GCN: self-loop weight 2) ----
  set_val_kernel<<<g1(ND), blk, 0, stream>>>(dinvd, 2.f, ND);
  set_val_kernel<<<g1(NP), blk, 0, stream>>>(dinvp, 2.f, NP);
  edge_deg_kernel<<<g1(ED), blk, 0, stream>>>(d_ei + ED, d_ew, dinvd, ED);
  edge_deg_kernel<<<g1(EP), blk, 0, stream>>>(p_ei + EP, p_ew, dinvp, EP);
  deg_to_dinv_kernel<<<g1(ND), blk, 0, stream>>>(dinvd, ND);
  deg_to_dinv_kernel<<<g1(NP), blk, 0, stream>>>(dinvp, NP);

  auto cvtA = [&](const float* src, int M, int K, int Mp, int Kp) {
    convert_pad_a_kernel<<<g1((size_t)Mp * (Kp >> 3)), blk, 0, stream>>>(src, abuf, M, K, Mp, Kp);
  };
  auto cvtW = [&](const float* src, int K, int N, int Np, int Kp) {
    transpose_convert_w_kernel<<<g1((size_t)Np * (Kp >> 3)), blk, 0, stream>>>(src, wtbuf, K, N, Np, Kp);
  };
  auto gemm = [&](const __bf16* A, int lda, const __bf16* Bt, int ldb,
                  float* C32, int ldc32, int o32, __bf16* C16, int ldc16, int o16,
                  int M, int N, int Kp, int epi, const float* bias,
                  const float* ga, const float* be, const float* me, const float* va) {
    dim3 grid((N + BN - 1) / BN, (M + BM - 1) / BM);
    gemm_bf16_wmma<<<grid, dim3(256), 0, stream>>>(A, lda, Bt, ldb, C32, ldc32, o32,
                                                   C16, ldc16, o16, M, N, Kp, epi,
                                                   bias, ga, be, me, va);
  };

  // ---- branch 1: dv -> enc_in[:, 0:300] (bf16) ----
  { int K = 300, Kp = ceilN(K, 32), Mp = ceilN(ND, 128), Np = ceilN(300, 128);
    cvtA(d_vecs, ND, K, Mp, Kp);
    cvtW(W_dvec, K, 300, Np, Kp);
    gemm(abuf, Kp, wtbuf, Kp, hbuf, 300, 0, nullptr, 0, 0, ND, 300, Kp, 0,
         nullptr, nullptr, nullptr, nullptr, nullptr);
    self_loop_init_kernel<<<g1((size_t)ND * 75), blk, 0, stream>>>(hbuf, dinvd, aggbuf, ND, 75);
    edge_aggregate_kernel<<<dim3(ED), dim3(128), 0, stream>>>(d_ei, d_ei + ED, d_ew, dinvd, hbuf, aggbuf, 75);
    gather_bias_leaky_bf16_kernel<<<g1((size_t)B * 300), blk, 0, stream>>>(aggbuf, d_index, b_dvec, enc_in, B, 300, 1344, 0);
  }
  // ---- enc_in[:, 300:1324] = p_embeddings ; zero pad 1324:1344 ----
  copy_cols_bf16_kernel<<<g1((size_t)B * 1024), blk, 0, stream>>>(p_emb, enc_in, B, 1024, 1344, 300);
  zero_cols_bf16_kernel<<<g1((size_t)B * 20), blk, 0, stream>>>(enc_in, B, 1344, 1324, 20);

  // ---- feature = leaky(BN(enc_in @ enc_W + b)): f32 -> d_out, bf16 -> fc1_in[:,0:512]
  { int K = 1324, Kp = ceilN(K, 32), Np = 512;
    cvtW(enc_W, K, 512, Np, Kp);
    gemm(enc_in, Kp, wtbuf, Kp, y_out + B, 512, 0, fc1_in, 2560, 0,
         B, 512, Kp, 1, enc_b, enc_g, enc_be, enc_m, enc_v);
  }

  // ---- branch 2: ec -> fc1_in[:, 512:1536] ----
  { int K = 1024, Kp = 1024, Mp = ceilN(ND, 128), Np = 1024;
    cvtA(d_ecfps, ND, K, Mp, Kp);
    cvtW(W_decfp, K, 1024, Np, Kp);
    gemm(abuf, Kp, wtbuf, Kp, hbuf, 1024, 0, nullptr, 0, 0, ND, 1024, Kp, 0,
         nullptr, nullptr, nullptr, nullptr, nullptr);
    self_loop_init_kernel<<<g1((size_t)ND * 256), blk, 0, stream>>>(hbuf, dinvd, aggbuf, ND, 256);
    edge_aggregate_kernel<<<dim3(ED), dim3(256), 0, stream>>>(d_ei, d_ei + ED, d_ew, dinvd, hbuf, aggbuf, 256);
    gather_bias_leaky_bf16_kernel<<<g1((size_t)B * 1024), blk, 0, stream>>>(aggbuf, d_index, b_decfp, fc1_in, B, 1024, 2560, 512);
  }

  // ---- branch 3: go -> fc1_in[:, 1536:2560] ----
  { int K = 2812, Kp = ceilN(K, 32), Mp = ceilN(NP, 128), Np = 1024;
    cvtA(p_gos, NP, K, Mp, Kp);
    cvtW(W_pgo, K, 1024, Np, Kp);
    gemm(abuf, Kp, wtbuf, Kp, hbuf, 1024, 0, nullptr, 0, 0, NP, 1024, Kp, 0,
         nullptr, nullptr, nullptr, nullptr, nullptr);
    self_loop_init_kernel<<<g1((size_t)NP * 256), blk, 0, stream>>>(hbuf, dinvp, aggbuf, NP, 256);
    edge_aggregate_kernel<<<dim3(EP), dim3(256), 0, stream>>>(p_ei, p_ei + EP, p_ew, dinvp, hbuf, aggbuf, 256);
    gather_bias_leaky_bf16_kernel<<<g1((size_t)B * 1024), blk, 0, stream>>>(aggbuf, p_index, b_pgo, fc1_in, B, 1024, 2560, 1536);
  }

  // ---- h = BN(leaky(fc1_in @ fc1_W + b)) -> fc1_out (f32) ----
  { int K = 2560, Kp = 2560, Np = 512;
    cvtW(fc1_W, K, 512, Np, Kp);
    gemm(fc1_in, Kp, wtbuf, Kp, fc1_out, 512, 0, nullptr, 0, 0,
         B, 512, Kp, 2, fc1_b, fc_g, fc_be, fc_m, fc_v);
  }

  // ---- y = relu(h @ fc2_W + fc2_b) ----
  fc2_relu_kernel<<<dim3((B + 7) / 8), dim3(256), 0, stream>>>(fc1_out, fc2_W, fc2_b, y_out, B, 512);
}
